// MambaSP_19430432047164
// MI455X (gfx1250) — matrix operand order repaired and checked by reading, not compiled, not run on hardware
//
#include <hip/hip_runtime.h>

#define B_   32
#define L_   4096
#define NL_  4
#define NS_  64
#define E_   2
#define KC_  4
#define SR_  4096.0f
#define T_   16
#define NC_  (L_ / T_)

typedef __attribute__((ext_vector_type(16))) _Float16 v16h;
typedef __attribute__((ext_vector_type(8)))  float    v8f;

union FragU { v16h h; unsigned u[8]; };
union PairU { _Float16 h[2]; unsigned u; };

__device__ __forceinline__ float silu_f(float x) {
    return x * (1.0f / (1.0f + __expf(-x)));
}

// One wave32 per batch element; all 4 layers fused, activations ping-pong in LDS.
// Chunked (SSD-style) exact reformulation of the selective scan:
//   stage 1: G_e = Chat_e (16x64) x Bhat_e^T (64x16)          -> 2 WMMA per e
//   stage 2: y_e = [tril(G_e) | Chat_e | 0] @ [v_e; h_e; 0]   -> 3 WMMA per e
// => 10 v_wmma_f32_16x16x32_f16 per chunk, no cross-lane permute reductions.
__global__ __launch_bounds__(32) void mamba_fused(
    const float* __restrict__ x,
    const float* __restrict__ W_in,
    const float* __restrict__ conv_w,
    const float* __restrict__ conv_b,
    const float* __restrict__ W_x,
    const float* __restrict__ W_dt,
    const float* __restrict__ b_dt,
    const float* __restrict__ A_log,
    const float* __restrict__ D_skip,
    const float* __restrict__ W_out,
    float* __restrict__ out)
{
    __shared__ float buf[2][L_];                       // activations ping-pong (32 KB)
    __shared__ __align__(16) _Float16 sB [E_][T_][66]; // Bhat rows (odd dword stride)
    __shared__ __align__(16) _Float16 sM [E_][T_][100];// [Gmask(16)|Chat(64)|0(16)] rows
    __shared__ __align__(16) _Float16 sVH[E_][96];     // [v(16); h(64); 0(16)] f16
    __shared__ __align__(16) float sWx[E_][129];
    __shared__ __align__(16) float sAp[E_][NS_];       // |A| = exp(A_log)
    __shared__ __align__(16) float sH [E_][NS_];       // carried SSM state (f32)
    __shared__ __align__(16) float sV [E_][T_];        // v = delta*u (f32, for carry)
    __shared__ __align__(16) float sY [E_][T_];

    const int b    = blockIdx.x;
    const int l    = threadIdx.x;
    const int t    = l & 15;    // time-in-chunk (and D-fragment column)
    const int half = l >> 4;    // == gate channel e for pointwise work
    const int e    = half;

    for (int i = l; i < L_; i += 32) buf[0][i] = x[b * L_ + i];
    __syncthreads();

    for (int layer = 0; layer < NL_; ++layer) {
        const float* in = buf[layer & 1];
        float*       ob = buf[(layer + 1) & 1];

        const float win_u = W_in[layer * 4 + e];
        const float win_z = W_in[layer * 4 + 2 + e];
        float cw[KC_];
        #pragma unroll
        for (int k = 0; k < KC_; ++k) cw[k] = conv_w[(layer * E_ + e) * KC_ + k];
        const float cb  = conv_b[layer * E_ + e];
        const float wdt = W_dt [layer * E_ + e];
        const float bdt = b_dt [layer * E_ + e];
        const float dsk = D_skip[layer * E_ + e];
        const float wo  = W_out [layer * E_ + e];

        for (int i = l; i < E_ * 129; i += 32)
            (&sWx[0][0])[i] = W_x[layer * E_ * 129 + i];
        for (int i = l; i < E_ * NS_; i += 32) {
            (&sAp[0][0])[i] = __expf(A_log[layer * E_ * NS_ + i]);
            (&sH [0][0])[i] = 0.0f;
        }
        // zero [v;h;0] vectors (h0 = 0) : 2*96 halves = 96 dwords
        for (int i = l; i < 96; i += 32) ((unsigned*)&sVH[0][0])[i] = 0u;
        // zero pad columns 80..95 of sM once per layer (8 dwords x 32 rows)
        for (int i = l; i < 256; i += 32) {
            const int ee = i >> 7, r = (i >> 3) & 15, cp = i & 7;
            *(unsigned*)&sM[ee][r][80 + 2 * cp] = 0u;
        }
        __syncthreads();

        for (int c = 0; c < NC_; ++c) {
            const int tg = c * T_ + t;

            // ---- x*W_in -> causal depthwise conv (u is linear in x) -> SiLU
            float acv = cb;
            #pragma unroll
            for (int k = 0; k < KC_; ++k) {
                const int s = tg - (KC_ - 1) + k;
                const float xv = (s >= 0) ? in[s] : 0.0f;
                acv = fmaf(xv * win_u, cw[k], acv);
            }
            const float u    = silu_f(acv);
            const float xres = in[tg];
            const float z    = xres * win_z;

            const float uo = __shfl_xor(u, 16, 32);
            const float u0 = half ? uo : u;
            const float u1 = half ? u  : uo;

            // ---- delta = softplus(dt*W_dt + b_dt)/SR
            float dtv = u0 * sWx[0][0] + u1 * sWx[1][0];
            dtv = dtv * wdt + bdt;
            const float spv   = (dtv > 20.0f) ? dtv : log1pf(__expf(dtv));
            const float delta = spv * (1.0f / SR_);

            // ---- inclusive prefix sum of delta within each 16-lane half
            float cum = delta;
            #pragma unroll
            for (int d = 1; d < 16; d <<= 1) {
                const float tmp = __shfl_up(cum, d, 16);
                if (t >= d) cum += tmp;
            }
            const float cumo  = __shfl_xor(cum, 16, 32);
            const float cume0 = half ? cumo : cum;
            const float cume1 = half ? cum  : cumo;

            const float vval = delta * u;
            sV[e][t]  = vval;
            sVH[e][t] = (_Float16)vval;

            // ---- rows: half 0 -> Bhat (both e); half 1 -> Chat into sM cols 16..79
            if (half == 0) {
                #pragma unroll 4
                for (int n = 0; n < NS_; n += 2) {
                    const float b0 = u0 * sWx[0][1 + n]     + u1 * sWx[1][1 + n];
                    const float b1 = u0 * sWx[0][1 + n + 1] + u1 * sWx[1][1 + n + 1];
                    PairU p0, p1;
                    p0.h[0] = (_Float16)(b0 * __expf(sAp[0][n]     * cume0));
                    p0.h[1] = (_Float16)(b1 * __expf(sAp[0][n + 1] * cume0));
                    p1.h[0] = (_Float16)(b0 * __expf(sAp[1][n]     * cume1));
                    p1.h[1] = (_Float16)(b1 * __expf(sAp[1][n + 1] * cume1));
                    *(unsigned*)&sB[0][t][n] = p0.u;
                    *(unsigned*)&sB[1][t][n] = p1.u;
                }
            } else {
                #pragma unroll 4
                for (int n = 0; n < NS_; n += 2) {
                    const float c0 = u0 * sWx[0][65 + n]     + u1 * sWx[1][65 + n];
                    const float c1 = u0 * sWx[0][65 + n + 1] + u1 * sWx[1][65 + n + 1];
                    PairU p0, p1;
                    p0.h[0] = (_Float16)(c0 * __expf(-sAp[0][n]     * cume0));
                    p0.h[1] = (_Float16)(c1 * __expf(-sAp[0][n + 1] * cume0));
                    p1.h[0] = (_Float16)(c0 * __expf(-sAp[1][n]     * cume1));
                    p1.h[1] = (_Float16)(c1 * __expf(-sAp[1][n + 1] * cume1));
                    *(unsigned*)&sM[0][t][16 + n] = p0.u;
                    *(unsigned*)&sM[1][t][16 + n] = p1.u;
                }
            }
            __syncthreads();

            // ---- stage 1: G_e = Chat_e x Bhat_e^T (K=64, 2 WMMA each),
            //      causal-masked D-fragment stored as f16 into sM cols 0..15
            #pragma unroll
            for (int ee = 0; ee < 2; ++ee) {
                v8f acc = {};
                #pragma unroll
                for (int sl = 0; sl < 2; ++sl) {
                    FragU fa, fb;
                    #pragma unroll
                    for (int v = 0; v < 8; ++v) {
                        const int ka = 16 + ((v >> 2) << 4) + ((v & 3) << 1)
                                     + (half << 3) + (sl << 5);
                        fa.u[v] = *(const unsigned*)&sM[ee][t][ka];
                        const int kb = ((v >> 2) << 4) + ((v & 3) << 1) + (half << 3) + (sl << 5);
                        fb.u[v] = *(const unsigned*)&sB[ee][t][kb];
                    }
                    acc = __builtin_amdgcn_wmma_f32_16x16x32_f16(
                        false, fa.h, false, fb.h, (short)0, acc, false, false);
                }
                #pragma unroll
                for (int v = 0; v < 8; ++v) {
                    const int trow = v + (half << 3);
                    const float mg = (t <= trow) ? acc[v] : 0.0f;  // causal mask
                    sM[ee][trow][t] = (_Float16)mg;
                }
            }
            __syncthreads();

            // ---- stage 2: y_e = [tril(G_e)|Chat_e|0] @ [v_e;h_e;0] (K=96, 3 WMMA),
            //      B operand replicated across all 16 columns -> no lane masking
            #pragma unroll
            for (int ee = 0; ee < 2; ++ee) {
                v8f acc = {};
                #pragma unroll
                for (int sl = 0; sl < 3; ++sl) {
                    FragU fa, fb;
                    #pragma unroll
                    for (int v = 0; v < 8; ++v) {
                        const int ka = ((v >> 2) << 4) + ((v & 3) << 1)
                                     + (half << 3) + (sl << 5);
                        fa.u[v] = *(const unsigned*)&sM[ee][t][ka];
                        const int kb = ((v >> 2) << 4) + ((v & 3) << 1) + (half << 3) + (sl << 5);
                        fb.u[v] = *(const unsigned*)&sVH[ee][kb];
                    }
                    acc = __builtin_amdgcn_wmma_f32_16x16x32_f16(
                        false, fa.h, false, fb.h, (short)0, acc, false, false);
                }
                if (t == 0) {   // column-0 lanes (0 and 16) hold y rows 0..7 / 8..15
                    #pragma unroll
                    for (int v = 0; v < 8; ++v) sY[ee][v + (half << 3)] = acc[v];
                }
            }
            __syncthreads();

            // ---- carry: h_out = exp(A*cumT) * (h_in + Bhat^T (delta*u))
            {
                const float cT = __shfl(cum, (half << 4) + 15, 32);
                const int nb = t << 2;
                float vs[T_];
                #pragma unroll
                for (int ss = 0; ss < T_; ++ss) vs[ss] = sV[half][ss];
                float ssum0 = 0.f, ssum1 = 0.f, ssum2 = 0.f, ssum3 = 0.f;
                #pragma unroll
                for (int ss = 0; ss < T_; ++ss) {
                    PairU pa, pb;
                    pa.u = *(const unsigned*)&sB[half][ss][nb];
                    pb.u = *(const unsigned*)&sB[half][ss][nb + 2];
                    ssum0 = fmaf((float)pa.h[0], vs[ss], ssum0);
                    ssum1 = fmaf((float)pa.h[1], vs[ss], ssum1);
                    ssum2 = fmaf((float)pb.h[0], vs[ss], ssum2);
                    ssum3 = fmaf((float)pb.h[1], vs[ss], ssum3);
                }
                float hn[4] = { ssum0, ssum1, ssum2, ssum3 };
                PairU q0, q1;
                #pragma unroll
                for (int j = 0; j < 4; ++j) {
                    const int n = nb + j;
                    const float hv = __expf(-sAp[half][n] * cT) * (sH[half][n] + hn[j]);
                    sH[half][n] = hv;
                    if (j < 2) q0.h[j] = (_Float16)hv; else q1.h[j - 2] = (_Float16)hv;
                }
                *(unsigned*)&sVH[half][16 + nb]     = q0.u;
                *(unsigned*)&sVH[half][16 + nb + 2] = q1.u;
            }
            __syncthreads();

            // ---- D-skip, z-gate, output projection, residual
            {
                float y = sY[e][t] + u * dsk;
                y *= silu_f(z);
                const float g  = y * wo;
                const float go = __shfl_xor(g, 16, 32);
                if (half == 0) ob[tg] = g + go + xres;
            }
            __syncthreads();
        } // chunks
        __syncthreads();
    } // layers

    const float* res = buf[NL_ & 1];
    for (int i = l; i < L_; i += 32) out[b * L_ + i] = res[i];
}

extern "C" void kernel_launch(void* const* d_in, const int* in_sizes, int n_in,
                              void* d_out, int out_size, void* d_ws, size_t ws_size,
                              hipStream_t stream) {
    (void)in_sizes; (void)n_in; (void)out_size; (void)d_ws; (void)ws_size;
    const float* x      = (const float*)d_in[0];
    const float* W_in   = (const float*)d_in[1];
    const float* conv_w = (const float*)d_in[2];
    const float* conv_b = (const float*)d_in[3];
    const float* W_x    = (const float*)d_in[4];
    const float* W_dt   = (const float*)d_in[5];
    const float* b_dt   = (const float*)d_in[6];
    const float* A_log  = (const float*)d_in[7];
    const float* D_skip = (const float*)d_in[8];
    const float* W_out  = (const float*)d_in[9];

    mamba_fused<<<B_, 32, 0, stream>>>(x, W_in, conv_w, conv_b, W_x, W_dt,
                                       b_dt, A_log, D_skip, W_out,
                                       (float*)d_out);
}
